// graph_norm_v_9216999817573
// MI455X (gfx1250) — matrix operand order repaired and checked by reading, not compiled, not run on hardware
//
#include <hip/hip_runtime.h>
#include <math.h>

typedef _Float16 half8  __attribute__((ext_vector_type(8)));
typedef _Float16 half16 __attribute__((ext_vector_type(16)));
typedef float    float8 __attribute__((ext_vector_type(8)));

#define MOD_EPS 1e-6f
#define GMAX 1024   // upper bound on num_graphs (actual G=32); scratch sized to this

// ---------------------------------------------------------------------------
// Kernel 1: feat[n,f] = ||feat_v[n,f,:] + 1e-8||  and per-graph sum of feat.
// Block = 256 threads (one feature each) x a chunk of nodes. Segments are
// contiguous runs, so we accumulate locally and emit ~1 float atomic/thread.
// ---------------------------------------------------------------------------
__global__ __launch_bounds__(256)
void k_feat(const float* __restrict__ feat_v, const int* __restrict__ seg,
            float* __restrict__ feat, float* __restrict__ gsum,
            int N, int F, int nodes_per_blk)
{
    int f  = blockIdx.y * 256 + threadIdx.x;
    if (f >= F) return;
    int n0 = blockIdx.x * nodes_per_blk;
    int n1 = n0 + nodes_per_blk; if (n1 > N) n1 = N;

    int cur = -1; float acc = 0.0f;
    for (int n = n0; n < n1; ++n) {
        int g = seg[n];
        const float* p = feat_v + ((size_t)n * F + f) * 3;
        float x0 = p[0] + 1e-8f, x1 = p[1] + 1e-8f, x2 = p[2] + 1e-8f;
        float ft = sqrtf(x0*x0 + x1*x1 + x2*x2);
        feat[(size_t)n * F + f] = ft;
        if (g != cur) {
            if (cur >= 0) atomicAdd(&gsum[(size_t)cur * F + f], acc);
            cur = g; acc = 0.0f;
        }
        acc += ft;
    }
    if (cur >= 0) atomicAdd(&gsum[(size_t)cur * F + f], acc);
}

// per-graph node counts (float so the means divide directly)
__global__ void k_cnt(const int* __restrict__ seg, float* __restrict__ cnt, int N)
{
    int n = blockIdx.x * blockDim.x + threadIdx.x;
    if (n < N) atomicAdd(&cnt[seg[n]], 1.0f);
}

// ---------------------------------------------------------------------------
// Kernel 2: per-graph sum of (feat - u2)^2 with u2 = (gsum/cnt)*alpha
// ---------------------------------------------------------------------------
__global__ __launch_bounds__(256)
void k_sigma(const float* __restrict__ feat, const int* __restrict__ seg,
             const float* __restrict__ gsum, const float* __restrict__ cnt,
             const float* __restrict__ alpha, float* __restrict__ gss,
             int N, int F, int nodes_per_blk)
{
    int f  = blockIdx.y * 256 + threadIdx.x;
    if (f >= F) return;
    int n0 = blockIdx.x * nodes_per_blk;
    int n1 = n0 + nodes_per_blk; if (n1 > N) n1 = N;
    float al = alpha[f];

    int cur = -1; float acc = 0.0f, u2 = 0.0f;
    for (int n = n0; n < n1; ++n) {
        int g = seg[n];
        if (g != cur) {
            if (cur >= 0) atomicAdd(&gss[(size_t)cur * F + f], acc);
            cur = g; acc = 0.0f;
            u2 = gsum[(size_t)g * F + f] / cnt[g] * al;
        }
        float d = feat[(size_t)n * F + f] - u2;
        acc += d * d;
    }
    if (cur >= 0) atomicAdd(&gss[(size_t)cur * F + f], acc);
}

// ---------------------------------------------------------------------------
// Kernel 3: y = (feat-u2)/(sqrt(sigma+eps)+eps)*gamma+beta  -> f16
// ---------------------------------------------------------------------------
__global__ __launch_bounds__(256)
void k_y(const float* __restrict__ feat, const int* __restrict__ seg,
         const float* __restrict__ gsum, const float* __restrict__ gss,
         const float* __restrict__ cnt, const float* __restrict__ alpha,
         const float* __restrict__ gamma, const float* __restrict__ beta,
         _Float16* __restrict__ y, int N, int F)
{
    int total = N * F;
    for (int idx = blockIdx.x * 256 + threadIdx.x; idx < total;
         idx += gridDim.x * 256) {
        int n = idx / F;
        int f = idx - n * F;
        int g = seg[n];
        float c  = cnt[g];
        float u2 = gsum[(size_t)g * F + f] / c * alpha[f];
        float sn = sqrtf(gss[(size_t)g * F + f] / c + MOD_EPS);
        float yv = (feat[idx] - u2) / (sn + MOD_EPS) * gamma[f] + beta[f];
        y[idx] = (_Float16)yv;
    }
}

// W (fp32, row-major [j,k]) -> f16, same layout (K-contiguous == B operand order)
__global__ __launch_bounds__(256)
void k_wcvt(const float* __restrict__ W, _Float16* __restrict__ Wh, int total)
{
    for (int i = blockIdx.x * 256 + threadIdx.x; i < total; i += gridDim.x * 256)
        Wh[i] = (_Float16)W[i];
}

// ---------------------------------------------------------------------------
// Kernel 4: WMMA GEMM  gate = y @ W^T + b, fused epilogue
//           out[n,j,c] = gate[n,j] * feat_v[n,j,c] / (feat[n,j]+zeta[j]+1e-8)
// Block: 256 threads = 8 waves, 128(M) x 128(N) tile.
// Wave: 32(M) x 64(N) = 2x4 f32 accumulators, K stepped by 32.
// Fragment layouts used by this kernel (wave32; A and B share one K order):
//   A 16x32 f16 : lane L -> row M = L&15; K = h*16 + 0..15 (one contiguous run)
//   B 32x16 f16 : lane L -> col  = L&15; K = h*16 + 0..15 (one contiguous run)
//   C 16x16 f32 : lane L -> col  = L&15; VGPR r -> row M = r + 8*h
// ---------------------------------------------------------------------------
__global__ __launch_bounds__(256)
void k_gemm(const _Float16* __restrict__ Y,    // [N,F]  row-major
            const _Float16* __restrict__ Wh,   // [F,F]  row-major (j,k)
            const float* __restrict__ bias,
            const float* __restrict__ zeta,
            const float* __restrict__ feat,
            const float* __restrict__ feat_v,
            float* __restrict__ out,
            int F)
{
    const int lane  = threadIdx.x & 31;
    const int wave  = threadIdx.x >> 5;
    const int l15   = lane & 15;
    const int h     = lane >> 4;
    const int M0    = blockIdx.x * 128 + (wave & 3) * 32;
    const int N0    = blockIdx.y * 128 + (wave >> 2) * 64;

    float8 acc[2][4] = {};

    const _Float16* arow0 = Y + (size_t)(M0      + l15) * F;
    const _Float16* arow1 = Y + (size_t)(M0 + 16 + l15) * F;
    const _Float16* brow[4];
#pragma unroll
    for (int ni = 0; ni < 4; ++ni)
        brow[ni] = Wh + (size_t)(N0 + ni * 16 + l15) * F + h * 16;

    for (int k0 = 0; k0 < F; k0 += 32) {
        // prefetch next K tile of A into cache (global_prefetch_b8)
        __builtin_prefetch(arow0 + k0 + 32, 0, 0);
        __builtin_prefetch(arow1 + k0 + 32, 0, 0);

        half8 a0lo = *(const half8*)(arow0 + k0 + h * 16);
        half8 a0hi = *(const half8*)(arow0 + k0 + h * 16 + 8);
        half8 a1lo = *(const half8*)(arow1 + k0 + h * 16);
        half8 a1hi = *(const half8*)(arow1 + k0 + h * 16 + 8);
        half16 a[2];
        a[0] = __builtin_shufflevector(a0lo, a0hi, 0,1,2,3,4,5,6,7,8,9,10,11,12,13,14,15);
        a[1] = __builtin_shufflevector(a1lo, a1hi, 0,1,2,3,4,5,6,7,8,9,10,11,12,13,14,15);

        half16 b[4];
#pragma unroll
        for (int ni = 0; ni < 4; ++ni)
            b[ni] = *(const half16*)(brow[ni] + k0);

#pragma unroll
        for (int mi = 0; mi < 2; ++mi)
#pragma unroll
            for (int ni = 0; ni < 4; ++ni)
                acc[mi][ni] = __builtin_amdgcn_wmma_f32_16x16x32_f16(
                    false, a[mi], false, b[ni], (short)0, acc[mi][ni],
                    false, false);
    }

    // fused epilogue: bias add + vector rescale, write out[n,j,0..2]
#pragma unroll
    for (int ni = 0; ni < 4; ++ni) {
        int   j  = N0 + ni * 16 + l15;
        float bj = bias[j];
        float zj = zeta[j] + 1e-8f;
#pragma unroll
        for (int mi = 0; mi < 2; ++mi) {
#pragma unroll
            for (int r = 0; r < 8; ++r) {
                int    node = M0 + mi * 16 + h * 8 + r;
                float  gate = acc[mi][ni][r] + bj;
                size_t e    = (size_t)node * F + j;
                float  s    = gate / (feat[e] + zj);
                const float* fv = feat_v + e * 3;
                float*       op = out    + e * 3;
                op[0] = s * fv[0];
                op[1] = s * fv[1];
                op[2] = s * fv[2];
            }
        }
    }
}

// ---------------------------------------------------------------------------
extern "C" void kernel_launch(void* const* d_in, const int* in_sizes, int n_in,
                              void* d_out, int out_size, void* d_ws, size_t ws_size,
                              hipStream_t stream)
{
    const float* feat_v = (const float*)d_in[0];
    const int*   seg    = (const int*)  d_in[1];
    // d_in[2] = num_graphs (device scalar) — not needed on host; seg indexes directly
    const float* alpha  = (const float*)d_in[3];
    const float* beta   = (const float*)d_in[4];
    const float* gamma  = (const float*)d_in[5];
    const float* zeta   = (const float*)d_in[6];
    const float* W      = (const float*)d_in[7];
    const float* bias   = (const float*)d_in[8];

    const int N = in_sizes[1];   // nodes (32000)
    const int F = in_sizes[3];   // features (1024)

    // workspace layout (256B-aligned slices)
    char*  ws  = (char*)d_ws;
    size_t off = 0;
    auto carve = [&](size_t bytes) {
        void* p = ws + off;
        off += (bytes + 255) & ~(size_t)255;
        return p;
    };
    float*     feat = (float*)    carve((size_t)N * F * sizeof(float));
    _Float16*  Yh   = (_Float16*) carve((size_t)N * F * sizeof(_Float16));
    _Float16*  Wh   = (_Float16*) carve((size_t)F * F * sizeof(_Float16));
    float*     gsum = (float*)    carve((size_t)GMAX * F * sizeof(float));
    float*     gss  = (float*)    carve((size_t)GMAX * F * sizeof(float));
    float*     cnt  = (float*)    carve((size_t)GMAX * sizeof(float));

    hipMemsetAsync(gsum, 0, (size_t)GMAX * F * sizeof(float), stream);
    hipMemsetAsync(gss,  0, (size_t)GMAX * F * sizeof(float), stream);
    hipMemsetAsync(cnt,  0, (size_t)GMAX * sizeof(float), stream);

    // W -> f16 (independent; overlaps nothing but cheap)
    k_wcvt<<<1024, 256, 0, stream>>>(W, Wh, F * F);

    // counts
    k_cnt<<<(N + 255) / 256, 256, 0, stream>>>(seg, cnt, N);

    // pass 1: norms + graph sums
    const int NPB = 128;
    dim3 g1((N + NPB - 1) / NPB, (F + 255) / 256);
    k_feat<<<g1, 256, 0, stream>>>(feat_v, seg, feat, gsum, N, F, NPB);

    // pass 2: variance sums
    k_sigma<<<g1, 256, 0, stream>>>(feat, seg, gsum, cnt, alpha, gss, N, F, NPB);

    // pass 3: normalized y in f16
    k_y<<<2048, 256, 0, stream>>>(feat, seg, gsum, gss, cnt, alpha, gamma, beta,
                                  Yh, N, F);

    // pass 4: WMMA GEMM + fused rescale epilogue
    dim3 gg(N / 128, F / 128);
    k_gemm<<<gg, 256, 0, stream>>>(Yh, Wh, bias, zeta, feat, feat_v,
                                   (float*)d_out, F);
}
